// PolarityAwareConv_83571473646104
// MI455X (gfx1250) — matrix operand order repaired and checked by reading, not compiled, not run on hardware
//
#include <hip/hip_runtime.h>

typedef __attribute__((ext_vector_type(16))) _Float16 v16h;
typedef __attribute__((ext_vector_type(8)))  float    v8f;

#define DIN  256
#define DHID 128
#define DRAW 16

// -------------------------------------------------------------------------
// pack_kernel: one block, 256 threads.
//  - W  = ew2 @ lew  (128x256), packed f16 in WMMA B-layout
//  - vb = eb2 @ lew  (256) f32
//  - w1, w2 (256x256) packed f16 in WMMA B-layout
// B-layout for 16-bit B (KxN per 32-K chunk): lane = (n&15) + 16*((kin>>3)&1),
// element = (kin&7) + 8*(kin>>4), stored as [(kc*16+tile)*32 + lane][16 f16].
// -------------------------------------------------------------------------
__global__ void pack_kernel(const float* __restrict__ ew2, const float* __restrict__ eb2,
                            const float* __restrict__ lew, const float* __restrict__ w1,
                            const float* __restrict__ w2,
                            _Float16* __restrict__ Wpk, _Float16* __restrict__ w1pk,
                            _Float16* __restrict__ w2pk, float* __restrict__ vb)
{
    const int t = threadIdx.x;
    {   // vb
        float s = 0.f;
        for (int j = 0; j < DHID; ++j) s = fmaf(eb2[j], lew[j * DIN + t], s);
        vb[t] = s;
    }
    for (int idx = t; idx < DHID * DIN; idx += 256) {   // Wpk
        const int k = idx >> 8, n = idx & 255;
        float s = 0.f;
        for (int j = 0; j < DHID; ++j) s = fmaf(ew2[k * DHID + j], lew[j * DIN + n], s);
        const int kc = k >> 5, kin = k & 31, tile = n >> 4;
        const int ln = (n & 15) + (((kin >> 3) & 1) ? 16 : 0);
        Wpk[((((kc * 16 + tile) * 32) + ln) << 4) + ((kin & 7) + ((kin >> 4) << 3))] = (_Float16)s;
    }
    for (int idx = t; idx < DIN * DIN; idx += 256) {    // w1pk / w2pk
        const int k = idx >> 8, n = idx & 255;
        const int kc = k >> 5, kin = k & 31, tile = n >> 4;
        const int ln = (n & 15) + (((kin >> 3) & 1) ? 16 : 0);
        const int o = ((((kc * 16 + tile) * 32) + ln) << 4) + ((kin & 7) + ((kin >> 4) << 3));
        w1pk[o] = (_Float16)w1[k * DIN + n];
        w2pk[o] = (_Float16)w2[k * DIN + n];
    }
}

// h_buf = x  (scatter target starts at x, so h = x + aggr for free)
__global__ void init_kernel(const float* __restrict__ x, float* __restrict__ hbuf, int n4)
{
    const int i = blockIdx.x * blockDim.x + threadIdx.x;
    if (i < n4) ((float4*)hbuf)[i] = ((const float4*)x)[i];
}

// -------------------------------------------------------------------------
// edge_kernel: 256 threads = 8 waves; each wave handles 16 edges.
//   1) each lane computes p*relu(raw@ew1+eb1) directly into A-layout regs
//   2) 16 N-tiles x 4 K-chunks of v_wmma_f32_16x16x32_f16 against Wpk
//   3) fused epilogue: + p*vb + leb, gather x[src], relu, atomic scatter
// Tail handling is wave-uniform so the hot path has no exec-mask churn.
// -------------------------------------------------------------------------
__global__ __launch_bounds__(256)
void edge_kernel(const float* __restrict__ x, const int* __restrict__ edge_index,
                 const float* __restrict__ edge_attr, const float* __restrict__ ew1,
                 const float* __restrict__ eb1, const _Float16* __restrict__ Wpk,
                 const float* __restrict__ vb, const float* __restrict__ leb,
                 float* __restrict__ hbuf, int E_)
{
    __shared__ float ew1s[DRAW * DHID];
    __shared__ float eb1s[DHID];
    __shared__ float pS[8][16];
    __shared__ int   srcS[8][16];
    __shared__ int   dstS[8][16];

    const int tid  = threadIdx.x;
    const int wave = tid >> 5;
    const int lane = tid & 31;

    for (int i = tid; i < DRAW * DHID; i += 256) ew1s[i] = ew1[i];
    if (tid < DHID) eb1s[tid] = eb1[tid];

    const int ebase = (blockIdx.x * 8 + wave) * 16;
    const int m  = lane & 15;
    const int e  = ebase + m;
    const int ec = (e < E_) ? e : (E_ - 1);

    const float* ap = edge_attr + ec * (1 + DRAW);
    const float p = fminf(fmaxf(ap[0], 0.f), 1.f) + 0.01f;
    float raw[DRAW];
#pragma unroll
    for (int j = 0; j < DRAW; ++j) raw[j] = ap[1 + j];

    if (lane < 16) {
        pS[wave][m]   = p;
        srcS[wave][m] = edge_index[ec];
        dstS[wave][m] = edge_index[E_ + ec];
    }
    __syncthreads();

    // ---- hidden -> A-matrix layout (16x32 f16 per K-chunk) ----
    const int koff0 = (lane < 16) ? 0 : 8;
    v16h a[4];
#pragma unroll
    for (int kc = 0; kc < 4; ++kc) {
        v16h av;
#pragma unroll
        for (int g = 0; g < 2; ++g) {
            const int kb = 32 * kc + koff0 + 16 * g;
#pragma unroll
            for (int u = 0; u < 8; ++u) {
                const int k = kb + u;
                float hv = eb1s[k];
#pragma unroll
                for (int j = 0; j < DRAW; ++j) hv = fmaf(raw[j], ew1s[j * DHID + k], hv);
                hv = fmaxf(hv, 0.f) * p;
                av[g * 8 + u] = (_Float16)hv;
            }
        }
        a[kc] = av;
    }

    float pr[8];
    const float* xp[8];
    float* hp[8];
    const int rb = (lane < 16) ? 0 : 8;
#pragma unroll
    for (int r = 0; r < 8; ++r) {
        pr[r] = pS[wave][rb + r];
        xp[r] = x    + (size_t)srcS[wave][rb + r] * DIN;
        hp[r] = hbuf + (size_t)dstS[wave][rb + r] * DIN;
    }

    const int nl = lane & 15;
    const _Float16* wbase = Wpk + (lane << 4);

    if (ebase + 16 <= E_) {
        // -------- fast path: full wave of 16 edges, no guards --------
#pragma unroll 4
        for (int tile = 0; tile < 16; ++tile) {
            v8f acc = {};
#pragma unroll
            for (int kc = 0; kc < 4; ++kc) {
                const v16h b = *(const v16h*)(wbase + (((kc * 16 + tile) * 32) << 4));
                acc = __builtin_amdgcn_wmma_f32_16x16x32_f16(false, a[kc], false, b,
                                                             (short)0, acc, false, false);
            }
            const int n = tile * 16 + nl;
            const float vbn  = vb[n];
            const float lebn = leb[n];
            float xv[8];
#pragma unroll
            for (int r = 0; r < 8; ++r) xv[r] = xp[r][n];
#pragma unroll
            for (int r = 0; r < 8; ++r) {
                const float gv = fmaxf(xv[r] + acc[r] + fmaf(pr[r], vbn, lebn), 0.f);
                unsafeAtomicAdd(hp[r] + n, gv);
            }
        }
    } else {
        // -------- tail path: per-row guards (rare) --------
        for (int tile = 0; tile < 16; ++tile) {
            v8f acc = {};
#pragma unroll
            for (int kc = 0; kc < 4; ++kc) {
                const v16h b = *(const v16h*)(wbase + (((kc * 16 + tile) * 32) << 4));
                acc = __builtin_amdgcn_wmma_f32_16x16x32_f16(false, a[kc], false, b,
                                                             (short)0, acc, false, false);
            }
            const int n = tile * 16 + nl;
            const float vbn  = vb[n];
            const float lebn = leb[n];
#pragma unroll
            for (int r = 0; r < 8; ++r) {
                if (ebase + rb + r < E_) {
                    const float gv = fmaxf(xp[r][n] + acc[r] + fmaf(pr[r], vbn, lebn), 0.f);
                    unsafeAtomicAdd(hp[r] + n, gv);
                }
            }
        }
    }
}

// -------------------------------------------------------------------------
// node_kernel: 128 threads = 4 waves, 16 rows per block.
//   GEMM1 (h@w1+b1) -> LDS -> LayerNorm -> relu -> GEMM2 (@w2+b2) -> out
// -------------------------------------------------------------------------
__global__ __launch_bounds__(128)
void node_kernel(const float* __restrict__ hbuf, const _Float16* __restrict__ w1pk,
                 const float* __restrict__ b1, const float* __restrict__ ln_g,
                 const float* __restrict__ ln_b, const _Float16* __restrict__ w2pk,
                 const float* __restrict__ b2, float* __restrict__ out, int Nn)
{
    __shared__ float h1s[16 * DIN];
    __shared__ float lngs[DIN], lnbs[DIN];
    __shared__ float ps[16][8], pss[16][8];
    __shared__ float mus[16], rsigs[16];

    const int tid  = threadIdx.x;
    const int wave = tid >> 5;
    const int lane = tid & 31;
    for (int i = tid; i < DIN; i += 128) { lngs[i] = ln_g[i]; lnbs[i] = ln_b[i]; }

    const int m0   = blockIdx.x * 16;
    const int rloc = lane & 15;
    int row = m0 + rloc; if (row >= Nn) row = Nn - 1;
    const int koff0 = (lane < 16) ? 0 : 8;
    const int rb    = (lane < 16) ? 0 : 8;

    // A1 = f16(h rows) in A-layout
    v16h a1[8];
#pragma unroll
    for (int kc = 0; kc < 8; ++kc) {
        v16h av;
#pragma unroll
        for (int g = 0; g < 2; ++g) {
            const float* s = hbuf + (size_t)row * DIN + 32 * kc + koff0 + 16 * g;
#pragma unroll
            for (int u = 0; u < 8; ++u) av[g * 8 + u] = (_Float16)s[u];
        }
        a1[kc] = av;
    }

    // GEMM1: each wave does 4 N-tiles
    const _Float16* w1base = w1pk + (lane << 4);
#pragma unroll
    for (int ti = 0; ti < 4; ++ti) {
        const int tile = wave * 4 + ti;
        v8f acc = {};
#pragma unroll
        for (int kc = 0; kc < 8; ++kc) {
            const v16h b = *(const v16h*)(w1base + (((kc * 16 + tile) * 32) << 4));
            acc = __builtin_amdgcn_wmma_f32_16x16x32_f16(false, a1[kc], false, b,
                                                         (short)0, acc, false, false);
        }
        const int n  = tile * 16 + (lane & 15);
        const float bn = b1[n];
#pragma unroll
        for (int r = 0; r < 8; ++r) h1s[(rb + r) * DIN + n] = acc[r] + bn;
    }
    __syncthreads();

    {   // parallel LN stats: 8 partial sums per row, then reduce
        const int lrow = tid >> 3, seg = tid & 7;
        float s = 0.f, ss = 0.f;
        const float* hr = h1s + lrow * DIN + seg * 32;
#pragma unroll
        for (int k = 0; k < 32; ++k) { const float v = hr[k]; s += v; ss += v * v; }
        ps[lrow][seg] = s; pss[lrow][seg] = ss;
    }
    __syncthreads();
    if (tid < 16) {
        float s = 0.f, ss = 0.f;
#pragma unroll
        for (int i = 0; i < 8; ++i) { s += ps[tid][i]; ss += pss[tid][i]; }
        const float mu  = s * (1.f / DIN);
        const float var = ss * (1.f / DIN) - mu * mu;
        mus[tid] = mu; rsigs[tid] = rsqrtf(var + 1e-5f);
    }
    __syncthreads();

    const float mu = mus[rloc], rs = rsigs[rloc];
    v16h a2[8];
#pragma unroll
    for (int kc = 0; kc < 8; ++kc) {
        v16h av;
#pragma unroll
        for (int g = 0; g < 2; ++g) {
#pragma unroll
            for (int u = 0; u < 8; ++u) {
                const int k = 32 * kc + koff0 + 16 * g + u;
                float v = fmaf((h1s[rloc * DIN + k] - mu) * rs, lngs[k], lnbs[k]);
                av[g * 8 + u] = (_Float16)fmaxf(v, 0.f);
            }
        }
        a2[kc] = av;
    }

    const bool full = (m0 + 16) <= Nn;
    const _Float16* w2base = w2pk + (lane << 4);
#pragma unroll
    for (int ti = 0; ti < 4; ++ti) {
        const int tile = wave * 4 + ti;
        v8f acc = {};
#pragma unroll
        for (int kc = 0; kc < 8; ++kc) {
            const v16h b = *(const v16h*)(w2base + (((kc * 16 + tile) * 32) << 4));
            acc = __builtin_amdgcn_wmma_f32_16x16x32_f16(false, a2[kc], false, b,
                                                         (short)0, acc, false, false);
        }
        const int n  = tile * 16 + (lane & 15);
        const float bn = b2[n];
        float* op = out + (size_t)(m0 + rb) * DIN + n;
        if (full) {
#pragma unroll
            for (int r = 0; r < 8; ++r) op[r * DIN] = acc[r] + bn;
        } else {
#pragma unroll
            for (int r = 0; r < 8; ++r)
                if (m0 + rb + r < Nn) op[r * DIN] = acc[r] + bn;
        }
    }
}

extern "C" void kernel_launch(void* const* d_in, const int* in_sizes, int n_in,
                              void* d_out, int out_size, void* d_ws, size_t ws_size,
                              hipStream_t stream) {
    const float* x      = (const float*)d_in[0];
    const int*   eidx   = (const int*)d_in[1];
    const float* eattr  = (const float*)d_in[2];
    const float* ew1    = (const float*)d_in[3];
    const float* eb1    = (const float*)d_in[4];
    const float* ew2    = (const float*)d_in[5];
    const float* eb2    = (const float*)d_in[6];
    const float* lew    = (const float*)d_in[7];
    const float* leb    = (const float*)d_in[8];
    const float* w1     = (const float*)d_in[9];
    const float* b1     = (const float*)d_in[10];
    const float* ln_g   = (const float*)d_in[11];
    const float* ln_b   = (const float*)d_in[12];
    const float* w2     = (const float*)d_in[13];
    const float* b2     = (const float*)d_in[14];
    float* outp = (float*)d_out;

    const int N = in_sizes[0] / DIN;
    const int E = in_sizes[2] / (1 + DRAW);

    char* ws = (char*)d_ws;
    float*     hbuf = (float*)ws;                      size_t off = (size_t)N * DIN * 4;
    _Float16*  Wpk  = (_Float16*)(ws + off);           off += (size_t)DHID * DIN * 2;
    _Float16*  w1pk = (_Float16*)(ws + off);           off += (size_t)DIN * DIN * 2;
    _Float16*  w2pk = (_Float16*)(ws + off);           off += (size_t)DIN * DIN * 2;
    float*     vb   = (float*)(ws + off);              off += (size_t)DIN * 4;
    (void)ws_size; (void)n_in; (void)out_size;

    pack_kernel<<<1, 256, 0, stream>>>(ew2, eb2, lew, w1, w2, Wpk, w1pk, w2pk, vb);

    const int n4 = N * DIN / 4;
    init_kernel<<<(n4 + 255) / 256, 256, 0, stream>>>(x, hbuf, n4);

    const int edgeBlocks = (E + 127) / 128;   // 8 waves * 16 edges
    edge_kernel<<<edgeBlocks, 256, 0, stream>>>(x, eidx, eattr, ew1, eb1, Wpk, vb, leb, hbuf, E);

    const int nodeBlocks = (N + 15) / 16;
    node_kernel<<<nodeBlocks, 128, 0, stream>>>(hbuf, w1pk, b1, ln_g, ln_b, w2pk, b2, outp, N);
}
